// ScoreModel_31009663877368
// MI455X (gfx1250) — matrix operand order repaired and checked by reading, not compiled, not run on hardware
//
#include <hip/hip_runtime.h>
#include <hip/hip_bf16.h>
#include <math.h>

// ---------------------------------------------------------------------------
// Equivariant graph attention layer for MI455X (gfx1250, wave32, WMMA).
// Edge MLPs (~47 GFLOP hot spot) run on v_wmma_f32_16x16x32_f16 with fp16
// activations/weights and fp32 accumulation. Edge tiles staged via the
// Tensor Data Mover when available. Everything else is bandwidth-trivial.
// ---------------------------------------------------------------------------

#define N_NODES 32768
#define E_EDGES 524288
#define NS 20
#define NTP 16
#define NK 8
#define NTPV 4
#define FC 64
#define EDGE_F 60

#define DEV_INLINE __device__ __attribute__((always_inline)) inline

typedef __attribute__((ext_vector_type(16))) _Float16 v16h;
typedef __attribute__((ext_vector_type(8)))  _Float16 v8h;
typedef __attribute__((ext_vector_type(8)))  float    v8f;
typedef __attribute__((ext_vector_type(4)))  unsigned int v4u;
typedef __attribute__((ext_vector_type(4)))  int  v4i;
typedef __attribute__((ext_vector_type(8)))  int  v8i;

// Tensor Data Mover availability (builtin arity differs across toolchains).
#if defined(__has_builtin)
#if __has_builtin(__builtin_amdgcn_tensor_load_to_lds) && \
    __has_builtin(__builtin_amdgcn_s_wait_tensorcnt)
#define HAVE_TDM 1
#endif
#if __has_builtin(__builtin_amdgcn_tanhf)
#define FAST_TANH 1
#endif
#endif

// Packed-weight fragment offsets (units: _Float16 elements).
// Each matrix: Ktiles(=2) x Ntiles fragments, 512 halves per fragment.
#define H_KW1 0
#define H_KW2 4096
#define H_KW3 8192
#define H_VW1 16384
#define H_VW2 20480
#define H_VW3 24576
#define H_TOTAL 45056

// ---------------------------------------------------------------------------
// Weight repack: fp32 row-major (K x Ncols) -> B-fragment order.
// Fragment (kt,nt): lane l holds col = nt*16 + (l&15),
//   K = kt*32 + (l>=16 ? 16 : 0) + e  for e in 0..15 (contiguous per lane).
// ---------------------------------------------------------------------------
__global__ void pack_weights_kernel(const float* __restrict__ W, int Kact,
                                    int Ntiles, int Ncols,
                                    _Float16* __restrict__ dst) {
  int tile = blockIdx.x;          // 0 .. Ktiles*Ntiles-1  (frag index)
  int lane = threadIdx.x;         // 0..31
  int nt = tile % Ntiles;
  int kt = tile / Ntiles;
  int col = nt * 16 + (lane & 15);
  int kb  = kt * 32 + ((lane >> 4) << 4);
  _Float16* p = dst + ((size_t)tile * 32 + lane) * 16;
#pragma unroll
  for (int e = 0; e < 16; ++e) {
    int kr = kb + e;
    float v = (kr < Kact) ? W[(size_t)kr * Ncols + col] : 0.0f;
    p[e] = (_Float16)v;
  }
}

// ---------------------------------------------------------------------------
// WMMA helpers (always_inline so InferAddressSpaces specializes LDS/global).
// ---------------------------------------------------------------------------
DEV_INLINE v8f wmma_step(v16h a, v16h b, v8f c) {
  // D = A(16x32 f16) * B(32x16 f16) + C(16x16 f32)
  return __builtin_amdgcn_wmma_f32_16x16x32_f16(
      /*neg_a=*/false, a, /*neg_b=*/false, b,
      /*c_mod=*/(short)0, c, /*reuse_a=*/false, /*reuse_b=*/false);
}

// A fragment from an LDS tile of 16 rows x 64 halves (row-major).
// Contiguous k order, as in packed B: lanes 0-15 row M=lane, K = k0+{0..15};
// lanes 16-31 same rows, K shifted by +16.  Two 16B ds_load_b128 per lane.
DEV_INLINE v16h load_a_frag(const _Float16* __restrict__ buf, int lane,
                            int k0) {
  int row = lane & 15;
  int kb = k0 + ((lane >> 4) << 4);
  const _Float16* p = buf + row * 64 + kb;
  v8h lo = *(const v8h*)(p);
  v8h hi = *(const v8h*)(p + 8);
  return __builtin_shufflevector(lo, hi, 0, 1, 2, 3, 4, 5, 6, 7, 8, 9, 10, 11,
                                 12, 13, 14, 15);
}

// B fragment: 32 contiguous bytes per lane from the packed (global) buffer.
DEV_INLINE v16h load_b_frag(const _Float16* __restrict__ pack, int frag,
                            int lane) {
  return *(const v16h*)(pack + ((size_t)frag * 32 + lane) * 16);
}

DEV_INLINE float gelu_tanh(float x) {
  float x3 = x * x * x;
  float t = 0.7978845608028654f * (x + 0.044715f * x3);
#ifdef FAST_TANH
  return 0.5f * x * (1.0f + __builtin_amdgcn_tanhf(t));  // v_tanh_f32
#else
  return 0.5f * x * (1.0f + tanhf(t));
#endif
}

// One hidden MLP layer: in 16x64 (f16 LDS) -> out 16x64 (f16 LDS), bias+GELU.
DEV_INLINE void mlp_hidden(const _Float16* __restrict__ inBuf,
                           const _Float16* __restrict__ wPack,
                           const float* __restrict__ bias,
                           _Float16* __restrict__ outBuf, int lane) {
  v16h a0 = load_a_frag(inBuf, lane, 0);
  v16h a1 = load_a_frag(inBuf, lane, 32);
  int col = lane & 15;
  int rb = (lane >> 4) << 3;
#pragma unroll
  for (int nt = 0; nt < 4; ++nt) {
    v8f c = {};
    c = wmma_step(a0, load_b_frag(wPack, 0 * 4 + nt, lane), c);
    c = wmma_step(a1, load_b_frag(wPack, 1 * 4 + nt, lane), c);
    float b = bias[nt * 16 + col];
#pragma unroll
    for (int r = 0; r < 8; ++r) {
      outBuf[(rb + r) * 64 + nt * 16 + col] = (_Float16)gelu_tanh(c[r] + b);
    }
  }
}

// Output-layer chunk: 2 n-tiles (32 cols) -> f32 LDS stage [16][36], bias only.
DEV_INLINE void mlp_out_chunk(const _Float16* __restrict__ inBuf,
                              const _Float16* __restrict__ wPack,
                              const float* __restrict__ bias, int Ntiles,
                              int chunk, float* __restrict__ stage, int lane) {
  v16h a0 = load_a_frag(inBuf, lane, 0);
  v16h a1 = load_a_frag(inBuf, lane, 32);
  int col = lane & 15;
  int rb = (lane >> 4) << 3;
#pragma unroll
  for (int ntL = 0; ntL < 2; ++ntL) {
    int nt = chunk * 2 + ntL;
    v8f c = {};
    c = wmma_step(a0, load_b_frag(wPack, 0 * Ntiles + nt, lane), c);
    c = wmma_step(a1, load_b_frag(wPack, 1 * Ntiles + nt, lane), c);
    float b = bias[nt * 16 + col];
#pragma unroll
    for (int r = 0; r < 8; ++r) {
      stage[(rb + r) * 36 + ntL * 16 + col] = c[r] + b;
    }
  }
}

DEV_INLINE void atomic_add_f32(float* p, float v) {
  __hip_atomic_fetch_add(p, v, __ATOMIC_RELAXED, __HIP_MEMORY_SCOPE_AGENT);
}

// ---------------------------------------------------------------------------
// Node transform: x = node_data @ W_in / sqrt(NS);  q = x @ W_q / sqrt(NTP)
// ---------------------------------------------------------------------------
__global__ void node_kernel(const float* __restrict__ nd,
                            const float* __restrict__ Win,
                            const float* __restrict__ Wq,
                            float* __restrict__ xb, float* __restrict__ qb) {
  int n = blockIdx.x * blockDim.x + threadIdx.x;
  if (n >= N_NODES) return;
  float nx[NS];
#pragma unroll
  for (int s = 0; s < NS; ++s) nx[s] = nd[(size_t)n * NS + s];
  const float inv_sqrt_ns = 0.22360679774997896f;  // 1/sqrt(20)
  float xv[NTP];
#pragma unroll
  for (int i = 0; i < NTP; ++i) {
    float a = 0.f;
#pragma unroll
    for (int s = 0; s < NS; ++s) a += nx[s] * Win[s * NTP + i];
    xv[i] = a * inv_sqrt_ns;
    xb[(size_t)n * NTP + i] = xv[i];
  }
#pragma unroll
  for (int j = 0; j < NK; ++j) {
    float a = 0.f;
#pragma unroll
    for (int i = 0; i < NTP; ++i) a += xv[i] * Wq[i * NK + j];
    qb[(size_t)n * NK + j] = a * 0.25f;  // 1/sqrt(16)
  }
}

// ---------------------------------------------------------------------------
// Init: amax=-inf, z=0, agg=0, stats=0
// ---------------------------------------------------------------------------
__global__ void init_kernel(float* __restrict__ amax, float* __restrict__ z,
                            float* __restrict__ agg, float* __restrict__ stats) {
  int t = blockIdx.x * blockDim.x + threadIdx.x;
  int stride = gridDim.x * blockDim.x;
  if (t < N_NODES) {
    amax[t] = -__builtin_inff();
    z[t] = 0.0f;
  }
  if (t < 64) stats[t] = 0.0f;
  for (int i = t; i < N_NODES * 28; i += stride) agg[i] = 0.0f;
}

// ---------------------------------------------------------------------------
// Edge kernel: per 16-edge tile per wave, run both MLPs via WMMA, then the
// small per-edge contractions (k, v0, v1) and attention logit a in VALU.
// ---------------------------------------------------------------------------
#define WAVES_PER_BLOCK 4

__global__ __launch_bounds__(32 * WAVES_PER_BLOCK) void edge_kernel(
    const float* __restrict__ edge_data, const int* __restrict__ ei,
    const float* __restrict__ esh, const _Float16* __restrict__ packW,
    const float* __restrict__ kb1, const float* __restrict__ kb2,
    const float* __restrict__ kb3, const float* __restrict__ vb1,
    const float* __restrict__ vb2, const float* __restrict__ vb3,
    const float* __restrict__ xbuf, const float* __restrict__ qbuf,
    const float* __restrict__ wdot, float* __restrict__ aOut,
    float* __restrict__ vOut) {
  __shared__ alignas(16) _Float16 s_in[WAVES_PER_BLOCK][16 * 64];
  __shared__ alignas(16) _Float16 s_h1[WAVES_PER_BLOCK][16 * 64];
  __shared__ alignas(16) _Float16 s_h2[WAVES_PER_BLOCK][16 * 64];
  __shared__ alignas(16) float s_stage[WAVES_PER_BLOCK][16 * 36];
  __shared__ float s_wdot[64];
#ifdef HAVE_TDM
  __shared__ alignas(16) float s_raw[WAVES_PER_BLOCK][16 * EDGE_F];
#endif

  int lane = threadIdx.x & 31;
  int w = threadIdx.x >> 5;
  int tile = blockIdx.x * WAVES_PER_BLOCK + w;
  int e0 = tile * 16;

  if (threadIdx.x < 64) s_wdot[threadIdx.x] = wdot[threadIdx.x];

  // Prefetch the next tile's edge features (speculative, L2).
  if (e0 + 16 < E_EDGES)
    __builtin_prefetch(edge_data + (size_t)(e0 + 16) * EDGE_F, 0, 1);

#ifdef HAVE_TDM
  // Stage the raw 16x60 fp32 tile via the Tensor Data Mover (one 2D
  // descriptor per wave), then convert to f16 with zero padding.
  {
    unsigned long long ga =
        (unsigned long long)(const void*)(edge_data + (size_t)e0 * EDGE_F);
    unsigned int lds_off =
        (unsigned int)(unsigned long long)(const void*)&s_raw[w][0];
    v4u g0;
    g0[0] = 1u;                                   // count=1, user mode
    g0[1] = lds_off;                              // D#.lds_addr
    g0[2] = (unsigned int)ga;                     // D#.global_addr lo
    g0[3] = (unsigned int)(ga >> 32) | (2u << 30);  // addr hi | type=2
    v8i g1;
    g1[0] = (int)(2u << 16);        // data_size=4B; no multicast/pad/iter
    g1[1] = (int)(60u << 16);       // tensor_dim0[15:0]=60
    g1[2] = (int)(16u << 16);       // tensor_dim0 hi=0 | tensor_dim1 lo=16
    g1[3] = (int)(60u << 16);       // tensor_dim1 hi=0 | tile_dim0=60
    g1[4] = 16;                     // tile_dim1=16, tile_dim2=0
    g1[5] = 60;                     // tensor_dim0_stride lo = 60 elements
    g1[6] = 0;                      // stride0 hi | stride1 lo
    g1[7] = 0;                      // stride1 hi
    v4i z4 = {};
#if __clang_major__ >= 23
    v8i z8 = {};
    __builtin_amdgcn_tensor_load_to_lds(g0, g1, z4, z4, z8, 0);
#else
    __builtin_amdgcn_tensor_load_to_lds(g0, g1, z4, z4, 0);
#endif
    __builtin_amdgcn_s_wait_tensorcnt(0);
  }
  for (int idx = lane; idx < 16 * 64; idx += 32) {
    int r = idx >> 6, c = idx & 63;
    float v = (c < EDGE_F) ? s_raw[w][r * EDGE_F + c] : 0.0f;
    s_in[w][idx] = (_Float16)v;
  }
#else
  // Stage input tile (16 x 60 fp32 -> 16 x 64 f16, zero-padded).
  for (int idx = lane; idx < 16 * 64; idx += 32) {
    int r = idx >> 6, c = idx & 63;
    float v = (c < EDGE_F) ? edge_data[(size_t)(e0 + r) * EDGE_F + c] : 0.0f;
    s_in[w][idx] = (_Float16)v;
  }
#endif

  // Per-lane edge registers (lanes 0-15 own edge e0+lane; hi lanes mirror).
  int eL = e0 + (lane & 15);
  int srcL = ei[eL];
  int dstL = ei[E_EDGES + eL];
  float xe[NTP];
#pragma unroll
  for (int i = 0; i < NTP; ++i) xe[i] = xbuf[(size_t)srcL * NTP + i];
  float sh0 = esh[(size_t)eL * 9 + 0];
  float sh1[3];
#pragma unroll
  for (int m = 0; m < 3; ++m) sh1[m] = esh[(size_t)eL * 9 + 1 + m];

  __syncthreads();

  // ---------------- k-MLP ----------------
  mlp_hidden(s_in[w], packW + H_KW1, kb1, s_h1[w], lane);
  __syncthreads();
  mlp_hidden(s_h1[w], packW + H_KW2, kb2, s_h2[w], lane);
  __syncthreads();

  float kacc[NK] = {};
#pragma unroll
  for (int chunk = 0; chunk < 4; ++chunk) {  // 128 cols / 32
    mlp_out_chunk(s_h2[w], packW + H_KW3, kb3, 8, chunk, s_stage[w], lane);
    __syncthreads();
    if (lane < 16) {
      const float* row = &s_stage[w][lane * 36];
#pragma unroll
      for (int c = 0; c < 32; ++c) {
        int cg = chunk * 32 + c;        // wk[e][i][j], cg = i*8 + j
        kacc[cg & 7] += xe[cg >> 3] * row[c];
      }
    }
    __syncthreads();
  }

  // k scale + attention logit a = q[dst] . W_dot . k / NK
  if (lane < 16) {
    float ks = sh0 * 0.25f;  // sh0 / sqrt(NTP)
#pragma unroll
    for (int j = 0; j < NK; ++j) kacc[j] *= ks;
    float acc = 0.f;
#pragma unroll
    for (int i = 0; i < NK; ++i) {
      float qi = qbuf[(size_t)dstL * NK + i];
      float t = 0.f;
#pragma unroll
      for (int j = 0; j < NK; ++j) t += s_wdot[i * NK + j] * kacc[j];
      acc += qi * t;
    }
    aOut[eL] = acc * 0.125f;  // / NK
  }
  __syncthreads();

  // ---------------- v-MLP ----------------
  mlp_hidden(s_in[w], packW + H_VW1, vb1, s_h1[w], lane);
  __syncthreads();
  mlp_hidden(s_h1[w], packW + H_VW2, vb2, s_h2[w], lane);
  __syncthreads();

  float v0acc[NTP] = {};
  float v1acc[NTPV] = {};
#pragma unroll
  for (int chunk = 0; chunk < 10; ++chunk) {  // 320 cols / 32
    mlp_out_chunk(s_h2[w], packW + H_VW3, vb3, 20, chunk, s_stage[w], lane);
    __syncthreads();
    if (lane < 16) {
      const float* row = &s_stage[w][lane * 36];
#pragma unroll
      for (int c = 0; c < 32; ++c) {
        int cg = chunk * 32 + c;
        float val = row[c];
        if (cg < NTP * NTP) {           // wv0[e][i][o], cg = i*16 + o
          v0acc[cg & 15] += xe[cg >> 4] * val;
        } else {                        // wv1[e][i][o4], t = i*4 + o4
          int t = cg - NTP * NTP;
          v1acc[t & 3] += xe[t >> 2] * val;
        }
      }
    }
    __syncthreads();
  }

  if (lane < 16) {
    float s0 = sh0 * 0.25f;
    float* vp = vOut + (size_t)eL * 28;
#pragma unroll
    for (int o = 0; o < NTP; ++o) vp[o] = v0acc[o] * s0;
#pragma unroll
    for (int o4 = 0; o4 < NTPV; ++o4)
#pragma unroll
      for (int m = 0; m < 3; ++m)
        vp[NTP + o4 * 3 + m] = v1acc[o4] * sh1[m] * 0.25f;
  }
}

// ---------------------------------------------------------------------------
// Segment max over dst (sign-split int/uint atomic trick; amax init = -inf).
// ---------------------------------------------------------------------------
__global__ void seg_max_kernel(const float* __restrict__ a,
                               const int* __restrict__ ei,
                               float* __restrict__ amax) {
  int e = blockIdx.x * blockDim.x + threadIdx.x;
  if (e >= E_EDGES) return;
  int d = ei[E_EDGES + e];
  float v = a[e];
  if (v >= 0.0f)
    atomicMax((int*)&amax[d], __float_as_int(v));
  else
    atomicMin((unsigned int*)&amax[d], __float_as_uint(v));
}

__global__ void seg_exp_kernel(const float* __restrict__ a,
                               const int* __restrict__ ei,
                               const float* __restrict__ amax,
                               float* __restrict__ ea, float* __restrict__ z) {
  int e = blockIdx.x * blockDim.x + threadIdx.x;
  if (e >= E_EDGES) return;
  int d = ei[E_EDGES + e];
  float m = amax[d];
  if (!(m > -1e37f)) m = 0.0f;  // non-finite guard (nodes with no edges)
  float v = __expf(a[e] - m);
  ea[e] = v;
  atomic_add_f32(&z[d], v);
}

__global__ void seg_agg_kernel(const float* __restrict__ ea,
                               const float* __restrict__ z,
                               const int* __restrict__ ei,
                               const float* __restrict__ vbuf,
                               float* __restrict__ agg) {
  int e = blockIdx.x * blockDim.x + threadIdx.x;
  if (e >= E_EDGES) return;
  int d = ei[E_EDGES + e];
  float wgt = ea[e] / z[d];
  const float* vp = vbuf + (size_t)e * 28;
  float* ap = agg + (size_t)d * 28;
#pragma unroll
  for (int c = 0; c < 28; ++c) atomic_add_f32(&ap[c], wgt * vp[c]);
}

// ---------------------------------------------------------------------------
// Output projection + residual; accumulate layernorm stats.
// stats[0..19]=sum(s), [20..39]=sum(s^2), [40..43]=sum(vec^2) per vector chan.
// ---------------------------------------------------------------------------
__global__ void out_proj_kernel(const float* __restrict__ agg,
                                const float* __restrict__ nd,
                                const float* __restrict__ Wout0,
                                const float* __restrict__ Wout1,
                                float* __restrict__ sb, float* __restrict__ vecb,
                                float* __restrict__ stats) {
  int n = blockIdx.x * blockDim.x + threadIdx.x;
  if (n >= N_NODES) return;
  float ag[28];
#pragma unroll
  for (int c = 0; c < 28; ++c) ag[c] = agg[(size_t)n * 28 + c];

  float sv[NS];
#pragma unroll
  for (int c = 0; c < NS; ++c) {
    float a = 0.f;
#pragma unroll
    for (int i = 0; i < NTP; ++i) a += ag[i] * Wout0[i * NS + c];
    sv[c] = a * 0.25f + nd[(size_t)n * NS + c];  // / sqrt(NTP) + residual
    sb[(size_t)n * NS + c] = sv[c];
  }
  float vv[NTPV * 3];
#pragma unroll
  for (int o = 0; o < NTPV; ++o)
#pragma unroll
    for (int m = 0; m < 3; ++m) {
      float a = 0.f;
#pragma unroll
      for (int c4 = 0; c4 < NTPV; ++c4)
        a += ag[NTP + c4 * 3 + m] * Wout1[c4 * NTPV + o];
      vv[o * 3 + m] = a * 0.5f;  // / sqrt(NTPV)
      vecb[(size_t)n * 12 + o * 3 + m] = vv[o * 3 + m];
    }
#pragma unroll
  for (int c = 0; c < NS; ++c) {
    atomic_add_f32(&stats[c], sv[c]);
    atomic_add_f32(&stats[NS + c], sv[c] * sv[c]);
  }
#pragma unroll
  for (int o = 0; o < NTPV; ++o) {
    float t = vv[o * 3] * vv[o * 3] + vv[o * 3 + 1] * vv[o * 3 + 1] +
              vv[o * 3 + 2] * vv[o * 3 + 2];
    atomic_add_f32(&stats[2 * NS + o], t);
  }
}

__global__ void norm_kernel(const float* __restrict__ sb,
                            const float* __restrict__ vecb,
                            const float* __restrict__ stats,
                            const float* __restrict__ gs,
                            const float* __restrict__ bs,
                            const float* __restrict__ gv,
                            float* __restrict__ out) {
  int n = blockIdx.x * blockDim.x + threadIdx.x;
  if (n >= N_NODES) return;
  const float invN = 1.0f / (float)N_NODES;
#pragma unroll
  for (int c = 0; c < NS; ++c) {
    float mu = stats[c] * invN;
    float var = stats[NS + c] * invN - mu * mu;
    float val = (sb[(size_t)n * NS + c] - mu) * rsqrtf(var + 1e-5f) * gs[c] +
                bs[c];
    out[(size_t)n * 32 + c] = val;
  }
#pragma unroll
  for (int o = 0; o < NTPV; ++o) {
    float vn = stats[2 * NS + o] * invN * (1.0f / 3.0f);
    float inv = rsqrtf(vn + 1e-5f) * gv[o];
#pragma unroll
    for (int m = 0; m < 3; ++m)
      out[(size_t)n * 32 + NS + o * 3 + m] = vecb[(size_t)n * 12 + o * 3 + m] * inv;
  }
}

// ---------------------------------------------------------------------------
extern "C" void kernel_launch(void* const* d_in, const int* in_sizes, int n_in,
                              void* d_out, int out_size, void* d_ws,
                              size_t ws_size, hipStream_t stream) {
  const float* node_data  = (const float*)d_in[0];
  const int*   edge_index = (const int*)d_in[1];
  const float* edge_data  = (const float*)d_in[2];
  const float* edge_sh    = (const float*)d_in[3];
  const float* W_in  = (const float*)d_in[4];
  const float* W_q   = (const float*)d_in[5];
  const float* k_w1  = (const float*)d_in[6];
  const float* k_b1  = (const float*)d_in[7];
  const float* k_w2  = (const float*)d_in[8];
  const float* k_b2  = (const float*)d_in[9];
  const float* k_w3  = (const float*)d_in[10];
  const float* k_b3  = (const float*)d_in[11];
  const float* v_w1  = (const float*)d_in[12];
  const float* v_b1  = (const float*)d_in[13];
  const float* v_w2  = (const float*)d_in[14];
  const float* v_b2  = (const float*)d_in[15];
  const float* v_w3  = (const float*)d_in[16];
  const float* v_b3  = (const float*)d_in[17];
  const float* w_dot = (const float*)d_in[18];
  const float* W_out0 = (const float*)d_in[19];
  const float* W_out1 = (const float*)d_in[20];
  const float* gamma_s = (const float*)d_in[21];
  const float* beta_s  = (const float*)d_in[22];
  const float* gamma_v = (const float*)d_in[23];
  float* out = (float*)d_out;
  (void)in_sizes; (void)n_in; (void)out_size; (void)ws_size;

  // Workspace carve-up (256 B aligned).
  char* ws = (char*)d_ws;
  size_t off = 0;
  auto take = [&](size_t bytes) -> void* {
    void* p = ws + off;
    off = (off + bytes + 255) & ~(size_t)255;
    return p;
  };
  _Float16* packW = (_Float16*)take((size_t)H_TOTAL * 2);
  float* xb   = (float*)take((size_t)N_NODES * NTP * 4);
  float* qb   = (float*)take((size_t)N_NODES * NK * 4);
  float* ab   = (float*)take((size_t)E_EDGES * 4);
  float* eab  = (float*)take((size_t)E_EDGES * 4);
  float* vb   = (float*)take((size_t)E_EDGES * 28 * 4);
  float* amax = (float*)take((size_t)N_NODES * 4);
  float* zb   = (float*)take((size_t)N_NODES * 4);
  float* aggb = (float*)take((size_t)N_NODES * 28 * 4);
  float* sb   = (float*)take((size_t)N_NODES * NS * 4);
  float* vecb = (float*)take((size_t)N_NODES * 12 * 4);
  float* stats = (float*)take(64 * 4);

  // 1) Repack weights into WMMA B-fragment layout (f16).
  pack_weights_kernel<<<8, 32, 0, stream>>>(k_w1, EDGE_F, 4, 64, packW + H_KW1);
  pack_weights_kernel<<<8, 32, 0, stream>>>(k_w2, FC, 4, 64, packW + H_KW2);
  pack_weights_kernel<<<16, 32, 0, stream>>>(k_w3, FC, 8, 128, packW + H_KW3);
  pack_weights_kernel<<<8, 32, 0, stream>>>(v_w1, EDGE_F, 4, 64, packW + H_VW1);
  pack_weights_kernel<<<8, 32, 0, stream>>>(v_w2, FC, 4, 64, packW + H_VW2);
  pack_weights_kernel<<<40, 32, 0, stream>>>(v_w3, FC, 20, 320, packW + H_VW3);

  // 2) Node transforms x, q.
  node_kernel<<<N_NODES / 256, 256, 0, stream>>>(node_data, W_in, W_q, xb, qb);

  // 3) Init reduction buffers.
  init_kernel<<<N_NODES / 256, 256, 0, stream>>>(amax, zb, aggb, stats);

  // 4) Edge MLPs + per-edge k/v/a (WMMA-heavy).
  {
    int tiles = E_EDGES / 16;                       // 32768
    int blocks = tiles / WAVES_PER_BLOCK;           // 8192
    edge_kernel<<<blocks, 32 * WAVES_PER_BLOCK, 0, stream>>>(
        edge_data, edge_index, edge_sh, packW, k_b1, k_b2, k_b3, v_b1, v_b2,
        v_b3, xb, qb, w_dot, ab, vb);
  }

  // 5) Segment softmax + aggregation.
  seg_max_kernel<<<E_EDGES / 256, 256, 0, stream>>>(ab, edge_index, amax);
  seg_exp_kernel<<<E_EDGES / 256, 256, 0, stream>>>(ab, edge_index, amax, eab,
                                                    zb);
  seg_agg_kernel<<<E_EDGES / 256, 256, 0, stream>>>(eab, zb, edge_index, vb,
                                                    aggb);

  // 6) Output projection + residual + stats; then normalize.
  out_proj_kernel<<<N_NODES / 256, 256, 0, stream>>>(aggb, node_data, W_out0,
                                                     W_out1, sb, vecb, stats);
  norm_kernel<<<N_NODES / 256, 256, 0, stream>>>(sb, vecb, stats, gamma_s,
                                                 beta_s, gamma_v, out);
}
